// Model_33174327394500
// MI455X (gfx1250) — matrix operand order repaired and checked by reading, not compiled, not run on hardware
//
#include <hip/hip_runtime.h>
#include <hip/hip_bf16.h>
#include <stdint.h>

#define N_NODES  40000
#define N_EDGES  640000
#define N_GRAPHS 1024
#define LN_EPS   1e-5f

typedef __attribute__((ext_vector_type(16))) _Float16     v16h;
typedef __attribute__((ext_vector_type(8)))  float        v8f;
typedef __attribute__((ext_vector_type(8)))  unsigned int v8u;

__device__ __forceinline__ unsigned int pack2_f16(float a, float b) {
  _Float16 ha = (_Float16)a, hb = (_Float16)b;
  unsigned short ua = __builtin_bit_cast(unsigned short, ha);
  unsigned short ub = __builtin_bit_cast(unsigned short, hb);
  return (unsigned int)ua | ((unsigned int)ub << 16);
}

__device__ __forceinline__ float2 unpack2_f16(unsigned int u) {
  unsigned short lo = (unsigned short)(u & 0xffffu);
  unsigned short hi = (unsigned short)(u >> 16);
  float a = (float)__builtin_bit_cast(_Float16, lo);
  float b = (float)__builtin_bit_cast(_Float16, hi);
  return make_float2(a, b);
}

__device__ __forceinline__ v16h frag_from(uint4 lo, uint4 hi) {
  v8u u = {lo.x, lo.y, lo.z, lo.w, hi.x, hi.y, hi.z, hi.w};
  return __builtin_bit_cast(v16h, u);
}

// ---------------------------------------------------------------- utilities
__global__ void zero_f32(float* __restrict__ p, int n) {
  int i = blockIdx.x * 256 + threadIdx.x;
  if (i < n) p[i] = 0.f;
}

// Pack a [nLayers][K][128] f32 (K-major) weight into WMMA B-fragment layout:
// dword index = ((kk*8 + nt)*32 + lane)*8 + v ; element (k,n) with
// k = kk*32 + 16*(v>>2) + 8*(lane>>4) + 2*(v&3) (+half), n = nt*16 + (lane&15).
__global__ void pack_wmma_b(const float* __restrict__ W, unsigned int* __restrict__ out,
                            int K, int nLayers) {
  int elems = (K >> 5) * 2048;  // (K/32) * 8 ntiles * 32 lanes * 8 dwords
  int idx = blockIdx.x * 256 + threadIdx.x;
  if (idx >= elems * nLayers) return;
  int layer = idx / elems;
  int r     = idx - layer * elems;
  int v     = r & 7;
  int lane  = (r >> 3) & 31;
  int nt    = (r >> 8) & 7;
  int kk    = r >> 11;
  int k = kk * 32 + ((v >> 2) << 4) + ((lane >> 4) << 3) + ((v & 3) << 1);
  int n = nt * 16 + (lane & 15);
  const float* Wl = W + (size_t)layer * K * 128;
  out[idx] = pack2_f16(Wl[(size_t)k * 128 + n], Wl[(size_t)(k + 1) * 128 + n]);
}

// h16 = pack_f16(x @ atoms_W + atoms_b)   ([N,64]@[64,128] -> [N][64] packed dwords)
// 128 threads = 2 nodes per block, one column-pair per thread.
__global__ void __launch_bounds__(128) node_embed(const float* __restrict__ x,
    const float* __restrict__ W, const float* __restrict__ b,
    unsigned int* __restrict__ h16) {
  __shared__ float sx[2][64];
  const int t     = threadIdx.x;
  const int node0 = blockIdx.x * 2;
  sx[t >> 6][t & 63] = x[(size_t)node0 * 64 + t];
  __syncthreads();
  const int ln = t >> 6;   // local node
  const int cp = t & 63;   // column pair
  float a0 = b[2 * cp], a1 = b[2 * cp + 1];
  #pragma unroll
  for (int k = 0; k < 64; ++k) {
    float xv = sx[ln][k];
    a0 += xv * W[k * 128 + 2 * cp];
    a1 += xv * W[k * 128 + 2 * cp + 1];
  }
  h16[(size_t)(node0 + ln) * 64 + cp] = pack2_f16(a0, a1);
}

// ----------------------------------------------- message pass (WMMA, K=384)
// block = 256 threads (8 waves), 64 edges per block. h stored as packed f16.
__global__ void __launch_bounds__(256) msg_pass(
    const unsigned int* __restrict__ h16,
    const int* __restrict__ srcIdx, const int* __restrict__ dstIdx,
    const float* __restrict__ eraw,
    const float* __restrict__ bondsW, const float* __restrict__ bondsB,
    const unsigned int* __restrict__ Bpack,
    const float* __restrict__ mbias, const float* __restrict__ lnw,
    const float* __restrict__ lnb, float* __restrict__ agg) {
  __shared__ int   s_src[64], s_dst[64];
  __shared__ float s_raw[64 * 16];
  __shared__ float s_bw[16 * 128];
  __shared__ __align__(16) unsigned int s_A[64 * 192];  // [64 edges][384 f16]
  __shared__ float s_C[64 * 129];                        // stride 129: conflict-free
  __shared__ float s_mu[64], s_rs[64];

  const int t  = threadIdx.x;
  const int e0 = blockIdx.x * 64;

  if (t < 64) { s_src[t] = srcIdx[e0 + t]; s_dst[t] = dstIdx[e0 + t]; }
  for (int i = t; i < 64 * 16; i += 256) s_raw[i] = eraw[(size_t)e0 * 16 + i];
  for (int i = t; i < 16 * 128; i += 256) s_bw[i] = bondsW[i];
  __syncthreads();

  // Stage A = [h[src] | h[dst] | e] as f16 (h already packed; e recomputed, K=16).
  for (int idx = t; idx < 64 * 192; idx += 256) {
    int edge = idx / 192;
    int c2   = idx - edge * 192;  // packed dword column (2 halves)
    unsigned int pk;
    if (c2 < 128) {
      int row = (c2 < 64) ? s_src[edge] : s_dst[edge];
      pk = h16[(size_t)row * 64 + (c2 & 63)];     // straight dword gather, no cvt
    } else {
      int col = (c2 - 128) << 1;
      float v0 = bondsB[col], v1 = bondsB[col + 1];
      const float* r = &s_raw[edge * 16];
      #pragma unroll
      for (int k = 0; k < 16; ++k) {
        float rv = r[k];
        v0 += rv * s_bw[k * 128 + col];
        v1 += rv * s_bw[k * 128 + col + 1];
      }
      pk = pack2_f16(v0, v1);
    }
    s_A[edge * 192 + c2] = pk;
  }
  __syncthreads();

  const int lane = t & 31;
  const int nt   = t >> 5;       // wave -> N-tile (8 tiles of 16 cols)
  const int lrow = lane & 15;
  const int lhi  = lane >> 4;
  const float bn = mbias[nt * 16 + lrow];

  for (int mt = 0; mt < 4; ++mt) {
    v8f acc = {};
    #pragma unroll
    for (int kk = 0; kk < 12; ++kk) {
      const int arow = mt * 16 + lrow;
      const int d0   = kk * 16 + lhi * 4;  // A layout: lane<16 K{0-7,16-23}, lane>=16 K{8-15,24-31}
      uint4 al = *(const uint4*)&s_A[arow * 192 + d0];
      uint4 ah = *(const uint4*)&s_A[arow * 192 + d0 + 8];
      const uint4* bp = (const uint4*)&Bpack[(size_t)((kk * 8 + nt) * 32 + lane) * 8];
      uint4 bl = bp[0], bh = bp[1];
      acc = __builtin_amdgcn_wmma_f32_16x16x32_f16(
          false, frag_from(al, ah), false, frag_from(bl, bh),
          (short)0, acc, false, false);
    }
    #pragma unroll
    for (int v = 0; v < 8; ++v) {
      int m = mt * 16 + v + lhi * 8;       // C/D layout: VGPR v -> M=v (+8 for hi lanes)
      s_C[m * 129 + nt * 16 + lrow] = acc[v] + bn;
    }
  }
  __syncthreads();

  if (t < 64) {
    float s = 0.f, s2 = 0.f;
    #pragma unroll 4
    for (int c = 0; c < 128; ++c) { float v = s_C[t * 129 + c]; s += v; s2 += v * v; }
    float mu  = s * (1.f / 128.f);
    float var = s2 * (1.f / 128.f) - mu * mu;
    s_mu[t] = mu;
    s_rs[t] = rsqrtf(var + LN_EPS);
  }
  __syncthreads();

  // LN -> tanh -> scatter-add to agg[dst] (f32 atomics, full precision)
  for (int idx = t; idx < 64 * 128; idx += 256) {
    int row = idx >> 7, c = idx & 127;
    float v = (s_C[row * 129 + c] - s_mu[row]) * s_rs[row] * lnw[c] + lnb[c];
    v = tanhf(v);
    __hip_atomic_fetch_add(&agg[(size_t)s_dst[row] * 128 + c], v,
                           __ATOMIC_RELAXED, __HIP_MEMORY_SCOPE_AGENT);
  }
}

// ----------------------------------------------- node update (WMMA, K=256)
__global__ void __launch_bounds__(256) node_update(
    const float* __restrict__ agg, const unsigned int* __restrict__ h16,
    const unsigned int* __restrict__ Bpack,
    const float* __restrict__ ubias, const float* __restrict__ lnw,
    const float* __restrict__ lnb, unsigned int* __restrict__ hout16) {
  __shared__ __align__(16) unsigned int s_A[64 * 128];  // [64 nodes][256 f16]
  __shared__ float s_C[64 * 129];
  __shared__ float s_mu[64], s_rs[64];
  const int t  = threadIdx.x;
  const int n0 = blockIdx.x * 64;

  for (int idx = t; idx < 64 * 128; idx += 256) {
    int rowl = idx >> 7, c2 = idx & 127;
    int node = n0 + rowl;
    unsigned int pk;
    if (c2 < 64) {
      const float* p = agg + (size_t)node * 128 + (c2 << 1);
      pk = pack2_f16(p[0], p[1]);
    } else {
      pk = h16[(size_t)node * 64 + (c2 & 63)];
    }
    s_A[rowl * 128 + c2] = pk;
  }
  __syncthreads();

  const int lane = t & 31;
  const int nt   = t >> 5;
  const int lrow = lane & 15;
  const int lhi  = lane >> 4;
  const float bn = ubias[nt * 16 + lrow];

  for (int mt = 0; mt < 4; ++mt) {
    v8f acc = {};
    #pragma unroll
    for (int kk = 0; kk < 8; ++kk) {
      const int arow = mt * 16 + lrow;
      const int d0   = kk * 16 + lhi * 4;
      uint4 al = *(const uint4*)&s_A[arow * 128 + d0];
      uint4 ah = *(const uint4*)&s_A[arow * 128 + d0 + 8];
      const uint4* bp = (const uint4*)&Bpack[(size_t)((kk * 8 + nt) * 32 + lane) * 8];
      uint4 bl = bp[0], bh = bp[1];
      acc = __builtin_amdgcn_wmma_f32_16x16x32_f16(
          false, frag_from(al, ah), false, frag_from(bl, bh),
          (short)0, acc, false, false);
    }
    #pragma unroll
    for (int v = 0; v < 8; ++v) {
      int m = mt * 16 + v + lhi * 8;
      s_C[m * 129 + nt * 16 + lrow] = acc[v] + bn;
    }
  }
  __syncthreads();

  if (t < 64) {
    float s = 0.f, s2 = 0.f;
    #pragma unroll 4
    for (int c = 0; c < 128; ++c) { float v = s_C[t * 129 + c]; s += v; s2 += v * v; }
    float mu  = s * (1.f / 128.f);
    float var = s2 * (1.f / 128.f) - mu * mu;
    s_mu[t] = mu;
    s_rs[t] = rsqrtf(var + LN_EPS);
  }
  __syncthreads();

  // LN -> tanh -> store h as packed f16 pairs
  for (int idx = t; idx < 64 * 64; idx += 256) {
    int row = idx >> 6, cp = idx & 63;
    int c0 = cp << 1, c1 = c0 + 1;
    float mu = s_mu[row], rs = s_rs[row];
    float v0 = tanhf((s_C[row * 129 + c0] - mu) * rs * lnw[c0] + lnb[c0]);
    float v1 = tanhf((s_C[row * 129 + c1] - mu) * rs * lnw[c1] + lnb[c1]);
    hout16[(size_t)(n0 + row) * 64 + cp] = pack2_f16(v0, v1);
  }
}

// ------------------------------- fc1 -> relu -> fc2 -> graph mean-pool (sums)
__global__ void __launch_bounds__(256) fc_pool(
    const unsigned int* __restrict__ h16, const int* __restrict__ batch,
    const float* __restrict__ W1, const float* __restrict__ b1,
    const float* __restrict__ W2, const float* __restrict__ b2,
    float* __restrict__ sums, float* __restrict__ cnt) {
  __shared__ float s_h[128];
  __shared__ float s_r[256];
  const int i = blockIdx.x, t = threadIdx.x;
  if (t < 64) {
    float2 hv = unpack2_f16(h16[(size_t)i * 64 + t]);
    s_h[2 * t]     = hv.x;
    s_h[2 * t + 1] = hv.y;
  }
  __syncthreads();
  float acc = b1[t];
  #pragma unroll 8
  for (int k = 0; k < 128; ++k) acc += s_h[k] * W1[k * 256 + t];
  acc = fmaxf(acc, 0.f) * W2[t];
  s_r[t] = acc;
  __syncthreads();
  for (int off = 128; off > 0; off >>= 1) {
    if (t < off) s_r[t] += s_r[t + off];
    __syncthreads();
  }
  if (t == 0) {
    float out = s_r[0] + b2[0];
    int g = batch[i];
    __hip_atomic_fetch_add(&sums[g], out, __ATOMIC_RELAXED, __HIP_MEMORY_SCOPE_AGENT);
    __hip_atomic_fetch_add(&cnt[g], 1.f, __ATOMIC_RELAXED, __HIP_MEMORY_SCOPE_AGENT);
  }
}

__global__ void finalize_pool(const float* __restrict__ sums, const float* __restrict__ cnt,
                              float* __restrict__ out) {
  int g = blockIdx.x * 256 + threadIdx.x;
  if (g < N_GRAPHS) out[g] = sums[g] / fmaxf(cnt[g], 1.f);
}

// ---------------------------------------------------------------- launcher
extern "C" void kernel_launch(void* const* d_in, const int* in_sizes, int n_in,
                              void* d_out, int out_size, void* d_ws, size_t ws_size,
                              hipStream_t stream) {
  (void)in_sizes; (void)n_in; (void)out_size; (void)ws_size;
  const float* x      = (const float*)d_in[0];
  const int*   eidx   = (const int*)d_in[1];
  const float* eraw   = (const float*)d_in[2];
  const int*   batch  = (const int*)d_in[3];
  const float* atomsW = (const float*)d_in[4];
  const float* atomsB = (const float*)d_in[5];
  const float* bondsW = (const float*)d_in[6];
  const float* bondsB = (const float*)d_in[7];
  const float* msgW   = (const float*)d_in[8];
  const float* msgB   = (const float*)d_in[9];
  const float* msgLnW = (const float*)d_in[10];
  const float* msgLnB = (const float*)d_in[11];
  const float* updW   = (const float*)d_in[12];
  const float* updB   = (const float*)d_in[13];
  const float* updLnW = (const float*)d_in[14];
  const float* updLnB = (const float*)d_in[15];
  const float* fc1W   = (const float*)d_in[16];
  const float* fc1B   = (const float*)d_in[17];
  const float* fc2W   = (const float*)d_in[18];
  const float* fc2B   = (const float*)d_in[19];
  float* out = (float*)d_out;

  char* ws = (char*)d_ws;
  size_t off = 0;
  auto carve = [&](size_t bytes) -> void* {
    void* p = ws + off;
    off += (bytes + 255) & ~(size_t)255;
    return p;
  };
  unsigned int* hA    = (unsigned int*)carve((size_t)N_NODES * 64 * 4);   // packed f16 h
  unsigned int* hB    = (unsigned int*)carve((size_t)N_NODES * 64 * 4);
  float*        agg   = (float*)carve((size_t)N_NODES * 128 * 4);
  unsigned int* msgPk = (unsigned int*)carve((size_t)3 * 24576 * 4);  // K=384: 12*8*32*8 dwords/layer
  unsigned int* updPk = (unsigned int*)carve((size_t)3 * 16384 * 4);  // K=256:  8*8*32*8 dwords/layer
  float*        sums  = (float*)carve((size_t)N_GRAPHS * 4);
  float*        cnt   = (float*)carve((size_t)N_GRAPHS * 4);

  const int* srcIdx = eidx;            // edge_index[0]
  const int* dstIdx = eidx + N_EDGES;  // edge_index[1]

  pack_wmma_b<<<(3 * 24576 + 255) / 256, 256, 0, stream>>>(msgW, msgPk, 384, 3);
  pack_wmma_b<<<(3 * 16384 + 255) / 256, 256, 0, stream>>>(updW, updPk, 256, 3);
  node_embed<<<N_NODES / 2, 128, 0, stream>>>(x, atomsW, atomsB, hA);

  unsigned int* hcur = hA; unsigned int* hnext = hB;
  for (int l = 0; l < 3; ++l) {
    zero_f32<<<(N_NODES * 128 + 255) / 256, 256, 0, stream>>>(agg, N_NODES * 128);
    msg_pass<<<N_EDGES / 64, 256, 0, stream>>>(hcur, srcIdx, dstIdx, eraw, bondsW, bondsB,
        msgPk + (size_t)l * 24576, msgB + l * 128, msgLnW + l * 128, msgLnB + l * 128, agg);
    node_update<<<N_NODES / 64, 256, 0, stream>>>(agg, hcur, updPk + (size_t)l * 16384,
        updB + l * 128, updLnW + l * 128, updLnB + l * 128, hnext);
    unsigned int* tmp = hcur; hcur = hnext; hnext = tmp;
  }

  zero_f32<<<(N_GRAPHS + 255) / 256, 256, 0, stream>>>(sums, N_GRAPHS);
  zero_f32<<<(N_GRAPHS + 255) / 256, 256, 0, stream>>>(cnt, N_GRAPHS);
  fc_pool<<<N_NODES, 256, 0, stream>>>(hcur, batch, fc1W, fc1B, fc2W, fc2B, sums, cnt);
  finalize_pool<<<(N_GRAPHS + 255) / 256, 256, 0, stream>>>(sums, cnt, out);
}
